// MultiHeadAttention_59571196395931
// MI455X (gfx1250) — matrix operand order repaired and checked by reading, not compiled, not run on hardware
//
#include <hip/hip_runtime.h>
#include <hip/hip_bf16.h>

// MI455X (gfx1250) wave32 WMMA causal multi-head attention.
// Pipeline: cast->bf16, fused QKV GEMM (V stored transposed) with async-to-LDS
// double-buffered weight staging, flash-attention (scores never touch HBM),
// projection GEMM + bias.

typedef __attribute__((ext_vector_type(16))) __bf16 v16bf;
typedef __attribute__((ext_vector_type(8)))  float  v8f;

#define BB  4
#define TT  2048
#define CC  1024
#define HH  16
#define HDD 64
#define MM  (BB * TT)      // 8192 rows of x

// ---------------------------------------------------------------------------
// Fragment loaders (CDNA5 16-bit WMMA layouts, wave32)
// A 16x32: lane (lr, kh) row=lr, halves K = kh*16 .. kh*16+15 (contiguous)
// B 32x16: lane (lr, kh) col=lr, halves K = kh*16 .. kh*16+15 (contiguous)
// ---------------------------------------------------------------------------
__device__ __forceinline__ v16bf load_frag_a(const __hip_bfloat16* rowp, int kh) {
    union { uint4 u[2]; v16bf v; } t;
    t.u[0] = *reinterpret_cast<const uint4*>(rowp + kh * 16);
    t.u[1] = *reinterpret_cast<const uint4*>(rowp + kh * 16 + 8);
    return t.v;
}
__device__ __forceinline__ v16bf load_frag_b(const __hip_bfloat16* rowp, int kh) {
    union { uint4 u[2]; v16bf v; } t;
    t.u[0] = *reinterpret_cast<const uint4*>(rowp + kh * 16);
    t.u[1] = *reinterpret_cast<const uint4*>(rowp + kh * 16 + 8);
    return t.v;
}
__device__ __forceinline__ v8f wmma_bf16(v16bf a, v16bf b, v8f c) {
    return __builtin_amdgcn_wmma_f32_16x16x32_bf16(false, a, false, b,
                                                   (short)0, c, false, false);
}

// ---------------------------------------------------------------------------
// f32 -> bf16 cast
// ---------------------------------------------------------------------------
__global__ void cast_f32_bf16(const float* __restrict__ in,
                              __hip_bfloat16* __restrict__ out, int n) {
    int i = blockIdx.x * blockDim.x + threadIdx.x;
    int stride = gridDim.x * blockDim.x;
    for (; i < n; i += stride) out[i] = __float2bfloat16(in[i]);
}

// ---------------------------------------------------------------------------
// Block-cooperative GEMM core: 256 threads (8 waves). Block tile = 256x64.
// Wave w computes rows [m0blk + 32w, +32) x cols [n0, n0+64).
// W slice (64 rows x 32 k) is staged into LDS with global_load_async_to_lds,
// double buffered; WMMAs for buffer b overlap the async transfer of b^1.
// A-fragment global loads are issued BEFORE the asynccnt wait + barrier so
// their latency hides under the block sync; all 4 B fragments are pulled from
// LDS before any WMMA so the ds waits can retire early.
// acc[su][j] covers rows su*16.., cols j*16.. of the wave tile.
// ---------------------------------------------------------------------------
__device__ __forceinline__ void gemm_core(const __hip_bfloat16* __restrict__ A,
                                          const __hip_bfloat16* __restrict__ W,
                                          int Kdim, int mw, int n0,
                                          __hip_bfloat16* lds,   // [2][64][32]
                                          v8f acc[2][4]) {
    const int tid  = threadIdx.x;
    const int lane = tid & 31, lr = lane & 15, kh = lane >> 4;

    // async staging: 256 threads x 16B cover 64 rows x 64B
    const int srow = tid >> 2;            // 0..63
    const int sseg = (tid & 3) * 8;       // half-element offset within row
    const __hip_bfloat16* wsrc = W + (size_t)(n0 + srow) * Kdim + sseg;
    // low 32 bits of a flat shared pointer == LDS byte offset (LDS aperture)
    const uint32_t ldst = (uint32_t)(uintptr_t)lds + (uint32_t)(srow * 32 + sseg) * 2u;
    const uint32_t bufbytes = 64 * 32 * 2;

    const __hip_bfloat16* arow0 = A + (size_t)(mw + lr) * Kdim;
    const __hip_bfloat16* arow1 = A + (size_t)(mw + 16 + lr) * Kdim;

    // prime buffer 0
    {
        uint64_t ga = (uint64_t)(uintptr_t)wsrc;
        asm volatile("global_load_async_to_lds_b128 %0, %1, off"
                     :: "v"(ldst), "v"(ga) : "memory");
    }
    int buf = 0;
    for (int k0 = 0; k0 < Kdim; k0 += 32, buf ^= 1) {
        // A fragments for this k-step: issue before the wait/barrier so the
        // loads are in flight while the block synchronizes.
        v16bf a0 = load_frag_a(arow0 + k0, kh);
        v16bf a1 = load_frag_a(arow1 + k0, kh);

        if (k0 + 32 < Kdim) {   // prefetch next slice into the other buffer
            uint64_t ga = (uint64_t)(uintptr_t)(wsrc + k0 + 32);
            uint32_t la = ldst + (uint32_t)(buf ^ 1) * bufbytes;
            asm volatile("global_load_async_to_lds_b128 %0, %1, off"
                         :: "v"(la), "v"(ga) : "memory");
            // async ops retire in order: <=1 outstanding => current buf landed
            asm volatile("s_wait_asynccnt 0x1" ::: "memory");
        } else {
            asm volatile("s_wait_asynccnt 0x0" ::: "memory");
        }
        __syncthreads();        // make staged slice visible block-wide

        // pull all 4 B fragments first so ds waits overlap WMMA issue
        const __hip_bfloat16* lb = lds + buf * (64 * 32);
        v16bf bfr[4];
#pragma unroll
        for (int j = 0; j < 4; ++j) {
            union { uint4 u[2]; v16bf v; } t;
            const __hip_bfloat16* brow = lb + (j * 16 + lr) * 32 + kh * 16;
            t.u[0] = *reinterpret_cast<const uint4*>(brow);
            t.u[1] = *reinterpret_cast<const uint4*>(brow + 8);
            bfr[j] = t.v;
        }
#pragma unroll
        for (int j = 0; j < 4; ++j) {
            acc[0][j] = wmma_bf16(a0, bfr[j], acc[0][j]);
            acc[1][j] = wmma_bf16(a1, bfr[j], acc[1][j]);
        }
        __syncthreads();        // all waves done with buf before it is rewritten
    }
}

// ---------------------------------------------------------------------------
// QKV GEMM: qkv = X @ Wqkv^T.
// Scatter: n in [0,C) -> Q [B,H,T,hd]; [C,2C) -> K [B,H,T,hd];
//          [2C,3C) -> V transposed [B,H,hd,T].
// ---------------------------------------------------------------------------
__global__ void qkv_gemm(const __hip_bfloat16* __restrict__ X,
                         const __hip_bfloat16* __restrict__ W,
                         __hip_bfloat16* __restrict__ Qc,
                         __hip_bfloat16* __restrict__ Kc,
                         __hip_bfloat16* __restrict__ Vt) {
    __shared__ __align__(16) __hip_bfloat16 lds[2][64][32];
    const int ntiles = (3 * CC) / 64;                 // 48
    const int n0 = (blockIdx.x % ntiles) * 64;
    const int m0 = (blockIdx.x / ntiles) * 256 + (threadIdx.x >> 5) * 32;
    const int lane = threadIdx.x & 31, lr = lane & 15, kh = lane >> 4;

    v8f acc[2][4] = {};
    gemm_core(X, W, CC, m0, n0, &lds[0][0][0], acc);

    const int which = n0 / CC;          // 0=Q, 1=K, 2=V
    const int h = (n0 % CC) / HDD;
    const int b = m0 / TT;
    const size_t hb = (size_t)(b * HH + h);
#pragma unroll
    for (int su = 0; su < 2; ++su) {
        const int t0 = (m0 % TT) + su * 16 + 8 * kh;
#pragma unroll
        for (int j = 0; j < 4; ++j) {
            const int c = j * 16 + lr;
#pragma unroll
            for (int v = 0; v < 8; ++v) {
                const int t = t0 + v;
                __hip_bfloat16 bv = __float2bfloat16(acc[su][j][v]);
                if (which == 0)      Qc[(hb * TT + t) * HDD + c] = bv;
                else if (which == 1) Kc[(hb * TT + t) * HDD + c] = bv;
                else                 Vt[(hb * HDD + c) * TT + t] = bv;
            }
        }
    }
}

// ---------------------------------------------------------------------------
// Projection GEMM: out = O @ Wproj^T + bias, f32 output.
// ---------------------------------------------------------------------------
__global__ void proj_gemm(const __hip_bfloat16* __restrict__ A,
                          const __hip_bfloat16* __restrict__ W,
                          const float* __restrict__ bias,
                          float* __restrict__ out) {
    __shared__ __align__(16) __hip_bfloat16 lds[2][64][32];
    const int ntiles = CC / 64;                       // 16
    const int n0 = (blockIdx.x % ntiles) * 64;
    const int m0 = (blockIdx.x / ntiles) * 256 + (threadIdx.x >> 5) * 32;
    const int lane = threadIdx.x & 31, lr = lane & 15, kh = lane >> 4;

    v8f acc[2][4] = {};
    gemm_core(A, W, CC, m0, n0, &lds[0][0][0], acc);

#pragma unroll
    for (int su = 0; su < 2; ++su) {
#pragma unroll
        for (int j = 0; j < 4; ++j) {
            const int col = n0 + j * 16 + lr;
            const float bv = bias[col];
#pragma unroll
            for (int v = 0; v < 8; ++v) {
                const int row = m0 + su * 16 + v + 8 * kh;
                out[(size_t)row * CC + col] = acc[su][j][v] + bv;
            }
        }
    }
}

// ---------------------------------------------------------------------------
// Flash attention: one wave per 16 query rows, key blocks of 32, online
// softmax.  P (f32 C-layout) is staged through wave-private LDS to become a
// bf16 A-fragment for the P@V WMMAs (V is pre-transposed -> contiguous B rows).
// V fragments are loaded right after the score WMMAs, so the softmax VALU
// block hides their global-load latency (they must be issued before the
// s_wait_dscnt fence, which the compiler may not hoist loads across).
// ---------------------------------------------------------------------------
__global__ void flash_attn(const __hip_bfloat16* __restrict__ Qc,
                           const __hip_bfloat16* __restrict__ Kc,
                           const __hip_bfloat16* __restrict__ Vt,
                           __hip_bfloat16* __restrict__ O) {
    __shared__ __align__(16) __hip_bfloat16 pbuf[8][16][40];  // 10 KB, +pad
    const float scale = 0.125f;  // hd^-0.5
    const int lane = threadIdx.x & 31, lr = lane & 15, kh = lane >> 4;
    const int w = threadIdx.x >> 5;
    const int qblocks = TT / 128;                      // 16
    const int bh = blockIdx.x / qblocks;
    const int q0 = (blockIdx.x % qblocks) * 128 + w * 16;

    const __hip_bfloat16* Qb = Qc + (size_t)bh * TT * HDD;
    const __hip_bfloat16* Kb = Kc + (size_t)bh * TT * HDD;
    const __hip_bfloat16* Vb = Vt + (size_t)bh * HDD * TT;

    const __hip_bfloat16* qrow = Qb + (size_t)(q0 + lr) * HDD;
    const v16bf aq0 = load_frag_a(qrow, kh);       // d = 0..31
    const v16bf aq1 = load_frag_a(qrow + 32, kh);  // d = 32..63

    float mstate[8], lstate[8];
#pragma unroll
    for (int v = 0; v < 8; ++v) { mstate[v] = -__builtin_inff(); lstate[v] = 0.f; }
    v8f oacc[4] = {};

    for (int kb = 0; kb < q0 + 16; kb += 32) {
        // score tiles: S = Q K^T for 32 keys (two 16-key column tiles)
        const __hip_bfloat16* krow0 = Kb + (size_t)(kb + lr) * HDD;
        const __hip_bfloat16* krow1 = Kb + (size_t)(kb + 16 + lr) * HDD;
        v16bf bk0a = load_frag_b(krow0, kh);
        v16bf bk0b = load_frag_b(krow0 + 32, kh);
        v16bf bk1a = load_frag_b(krow1, kh);
        v16bf bk1b = load_frag_b(krow1 + 32, kh);
        v8f s0 = {}, s1 = {};
        s0 = wmma_bf16(aq0, bk0a, s0);
        s0 = wmma_bf16(aq1, bk0b, s0);
        s1 = wmma_bf16(aq0, bk1a, s1);
        s1 = wmma_bf16(aq1, bk1b, s1);

        // V fragments for this key block: independent of softmax, issue now so
        // the ~60-op softmax VALU block below covers their latency.
        v16bf bv[4];
#pragma unroll
        for (int j = 0; j < 4; ++j)
            bv[j] = load_frag_b(Vb + (size_t)(j * 16 + lr) * TT + kb, kh);

        float alpha[8];
#pragma unroll
        for (int v = 0; v < 8; ++v) {
            const int row = q0 + v + 8 * kh;          // C-layout row
            const int c0 = kb + lr, c1 = kb + 16 + lr;
            float x0 = (c0 <= row) ? s0[v] * scale : -__builtin_inff();
            float x1 = (c1 <= row) ? s1[v] * scale : -__builtin_inff();
            float mx = fmaxf(x0, x1);
#pragma unroll
            for (int off = 1; off < 16; off <<= 1)
                mx = fmaxf(mx, __shfl_xor(mx, off, 32));
            const float mnew = fmaxf(mstate[v], mx);
            const float al = __expf(mstate[v] - mnew);
            const float e0 = __expf(x0 - mnew);
            const float e1 = __expf(x1 - mnew);
            float rs = e0 + e1;
#pragma unroll
            for (int off = 1; off < 16; off <<= 1)
                rs += __shfl_xor(rs, off, 32);
            lstate[v] = lstate[v] * al + rs;
            mstate[v] = mnew;
            alpha[v] = al;
            // stage P into LDS (bf16) for A-fragment reload
            pbuf[w][v + 8 * kh][lr]      = __float2bfloat16(e0);
            pbuf[w][v + 8 * kh][16 + lr] = __float2bfloat16(e1);
        }
#pragma unroll
        for (int j = 0; j < 4; ++j)
#pragma unroll
            for (int v = 0; v < 8; ++v) oacc[j][v] *= alpha[v];

        asm volatile("s_wait_dscnt 0x0" ::: "memory");   // wave-private LDS RAW
        union { uint4 u[2]; v16bf v; } pt;
        pt.u[0] = *reinterpret_cast<const uint4*>(&pbuf[w][lr][kh * 16]);
        pt.u[1] = *reinterpret_cast<const uint4*>(&pbuf[w][lr][kh * 16 + 8]);
        const v16bf ap = pt.v;

#pragma unroll
        for (int j = 0; j < 4; ++j)
            oacc[j] = wmma_bf16(ap, bv[j], oacc[j]);
    }

    // normalize and write bf16 attention output [B,T,C]
    const int b = bh / HH, h = bh % HH;
#pragma unroll
    for (int v = 0; v < 8; ++v) {
        const float inv = 1.0f / lstate[v];
        const int t = q0 + v + 8 * kh;
        const size_t rowbase = ((size_t)b * TT + t) * CC + h * HDD;
#pragma unroll
        for (int j = 0; j < 4; ++j)
            O[rowbase + j * 16 + lr] = __float2bfloat16(oacc[j][v] * inv);
    }
}

// ---------------------------------------------------------------------------
extern "C" void kernel_launch(void* const* d_in, const int* in_sizes, int n_in,
                              void* d_out, int out_size, void* d_ws, size_t ws_size,
                              hipStream_t stream) {
    (void)in_sizes; (void)n_in; (void)out_size; (void)ws_size;
    const float* x      = (const float*)d_in[0];
    const float* w_qkv  = (const float*)d_in[1];
    const float* w_proj = (const float*)d_in[2];
    const float* b_proj = (const float*)d_in[3];
    float* out = (float*)d_out;

    char* ws = (char*)d_ws;
    __hip_bfloat16* xb     = (__hip_bfloat16*)ws; ws += (size_t)MM * CC * 2;
    __hip_bfloat16* wqkvb  = (__hip_bfloat16*)ws; ws += (size_t)3 * CC * CC * 2;
    __hip_bfloat16* wprojb = (__hip_bfloat16*)ws; ws += (size_t)CC * CC * 2;
    __hip_bfloat16* Qc     = (__hip_bfloat16*)ws; ws += (size_t)MM * CC * 2;
    __hip_bfloat16* Kc     = (__hip_bfloat16*)ws; ws += (size_t)MM * CC * 2;
    __hip_bfloat16* Vtb    = (__hip_bfloat16*)ws; ws += (size_t)MM * CC * 2;
    __hip_bfloat16* Ob     = (__hip_bfloat16*)ws; ws += (size_t)MM * CC * 2;

    cast_f32_bf16<<<2048, 256, 0, stream>>>(x, xb, MM * CC);
    cast_f32_bf16<<<1024, 256, 0, stream>>>(w_qkv, wqkvb, 3 * CC * CC);
    cast_f32_bf16<<<512, 256, 0, stream>>>(w_proj, wprojb, CC * CC);

    // (M/256)*(3C/64) = 32*48 = 1536 blocks, 256 threads (8 waves)
    qkv_gemm<<<1536, 256, 0, stream>>>(xb, wqkvb, Qc, Kc, Vtb);
    // B*H * (T/128) = 1024 blocks, 8 waves each own 16 query rows
    flash_attn<<<1024, 256, 0, stream>>>(Qc, Kc, Vtb, Ob);
    // (M/256)*(C/64) = 32*16 = 512 blocks
    proj_gemm<<<512, 256, 0, stream>>>(Ob, wprojb, b_proj, out);
}
